// ResidualAdapter_81999515615896
// MI455X (gfx1250) — matrix operand order repaired and checked by reading, not compiled, not run on hardware
//
#include <hip/hip_runtime.h>
#include <hip/hip_bf16.h>

// ---------------------------------------------------------------------------
// ResidualAdapter forward for MI455X (gfx1250, wave32, WMMA + TDM).
// B=32, T=2048, H=64, S=256, G=256, NL=2, SL=5.
// Heavy GEMMs (x-projections): v_wmma_f32_16x16x32_bf16, operands staged to
// LDS by the Tensor Data Mover (double-buffered A tiles, padded rows for
// conflict-free ds_load_b128). Recurrent scans: persistent workgroup with
// register-resident WMMA B-fragments and software-pipelined xp loads.
// ---------------------------------------------------------------------------

#define Bt 32
#define Tt 2048
#define Ht 64
#define St 256
#define Gt 256

typedef __bf16 v16bf __attribute__((ext_vector_type(16)));
typedef float  v8f   __attribute__((ext_vector_type(8)));
typedef unsigned int u32x4  __attribute__((ext_vector_type(4)));
typedef int          i32x8  __attribute__((ext_vector_type(8)));
typedef int          i32x4  __attribute__((ext_vector_type(4)));
typedef float        f32x4  __attribute__((ext_vector_type(4)));
typedef __bf16       bf16x8 __attribute__((ext_vector_type(8)));

#if defined(__has_builtin)
#if __has_builtin(__builtin_amdgcn_tensor_load_to_lds)
#define HAVE_TDM 1
#endif
#endif

__device__ __forceinline__ void wait_tensorcnt0() {
#if defined(__has_builtin) && __has_builtin(__builtin_amdgcn_s_wait_tensorcnt)
  __builtin_amdgcn_s_wait_tensorcnt(0);
#else
  asm volatile("s_wait_tensorcnt 0x0" ::: "memory");
#endif
}
__device__ __forceinline__ void wait_tensorcnt1() {
#if defined(__has_builtin) && __has_builtin(__builtin_amdgcn_s_wait_tensorcnt)
  __builtin_amdgcn_s_wait_tensorcnt(1);
#else
  asm volatile("s_wait_tensorcnt 0x1" ::: "memory");
#endif
}

#ifdef HAVE_TDM
// 2-D tile load via Tensor Data Mover: rows x width bf16 elements, global row
// stride strideElems, into LDS at ldsOff with per-row padding codes
// (padInterval/padAmount per D# group1). D# layout per CDNA5 ISA.
// 6-arg builtin form (clang-23 / therock): (g0, g1, g2, g3, g3b, cpol).
__device__ __forceinline__ void tdm_load_2d(unsigned ldsOff, const void* gptr,
                                            unsigned rows, unsigned width,
                                            unsigned long long strideElems,
                                            unsigned padInterval, unsigned padAmount) {
  unsigned long long ga = (unsigned long long)gptr;
  u32x4 g0;
  g0[0] = 1u;                                   // count=1, no gather
  g0[1] = ldsOff;                               // lds_addr
  g0[2] = (unsigned)(ga & 0xFFFFFFFFu);         // global_addr[31:0]
  g0[3] = (unsigned)((ga >> 32) & 0x01FFFFFFu) | (2u << 30);  // addr[56:32]|type=2
  unsigned td0 = (unsigned)strideElems;         // tensor width (>= tile width)
  i32x8 g1;
  g1[0] = (int)((1u << 16) |                    // data_size=1 (2 bytes)
                (1u << 20) |                    // pad_enable
                (padInterval << 22) | (padAmount << 25));
  g1[1] = (int)(td0 << 16);                     // tensor_dim0 lo16 (ab_addr=0)
  g1[2] = (int)((td0 >> 16) | (rows << 16));    // tensor_dim0 hi16 | tensor_dim1 lo16
  g1[3] = (int)(width << 16);                   // tensor_dim1 hi16=0 | tile_dim0
  g1[4] = (int)(rows & 0xFFFFu);                // tile_dim1 | tile_dim2=0
  g1[5] = (int)(unsigned)strideElems;           // tensor_dim0_stride lo32
  g1[6] = (int)((strideElems >> 32) & 0xFFFFu); // stride hi16 | dim1_stride lo=0
  g1[7] = 0;
  i32x4 gz = {0, 0, 0, 0};
  i32x8 gz8 = {0, 0, 0, 0, 0, 0, 0, 0};
  __builtin_amdgcn_tensor_load_to_lds(g0, g1, gz, gz, gz8, 0);
}
#endif

__device__ __forceinline__ float sigf(float x) { return 1.0f / (1.0f + expf(-x)); }

// A-fragment (16-bit, 16x32) K mapping: dword j of lane half holds k = 16*laneHi + 2j (contiguous, matches B).
__device__ __forceinline__ int a_frag_k(int j, int laneHi) {
  return (laneHi << 4) + 2 * j;
}

#define WPAD 72  // padded W row stride (bf16 elems): 144B, conflict-free b128
#define APAD 36  // padded A row stride (bf16 elems): 72B

// one 16x32 k-step: A-frag from ldsA (padded rows), 16 B-frags from ldsW chunk
__device__ __forceinline__ void gemm_kstep(const __bf16* ldsA, const __bf16* ldsW,
                                           int ksOff, int wv, int lanelo, int laneHi,
                                           v8f acc[16]) {
  union { v16bf v; unsigned u[8]; } aF;
#pragma unroll
  for (int j = 0; j < 8; ++j)
    aF.u[j] = *(const unsigned*)(ldsA + (wv * 16 + lanelo) * APAD + a_frag_k(j, laneHi));
#pragma unroll
  for (int nt = 0; nt < 16; ++nt) {
    union { v16bf v; u32x4 q[2]; } bF;
    const __bf16* wr = ldsW + (nt * 16 + lanelo) * WPAD + ksOff + laneHi * 16;
    bF.q[0] = *(const u32x4*)wr;
    bF.q[1] = *(const u32x4*)(wr + 8);
    acc[nt] = __builtin_amdgcn_wmma_f32_16x16x32_bf16(
        false, aF.v, false, bF.v, (short)0, acc[nt], false, false);
  }
}

// ---------------------------------------------------------------------------
// out[M,256] = A[M,K](bf16) * W[256,K]^T + bias. grid=(M/128, nBatch), 8 waves.
// K in 64-wide chunks; TDM stages W chunk + double-buffered 128x32 A tiles.
// ---------------------------------------------------------------------------
__global__ __launch_bounds__(256)
void k_gemm_bf16(const __bf16* __restrict__ A, long long aBatch,
                 const __bf16* __restrict__ W, long long wBatch,
                 const float* __restrict__ bias, long long biasBatch,
                 float* __restrict__ out, long long outBatch,
                 int M, int K) {
  extern __shared__ char smem[];
  __bf16* ldsW  = (__bf16*)smem;                         // 256 x WPAD (36 KB)
  __bf16* ldsA0 = (__bf16*)(smem + 256 * WPAD * 2);      // 128 x APAD (9 KB)
  __bf16* ldsA1 = ldsA0 + 128 * APAD;                    // 128 x APAD (9 KB)

  const int tid = threadIdx.x;
  const int lane = tid & 31, wv = tid >> 5;
  const int laneHi = lane >> 4, lanelo = lane & 15;
  const long long batch = blockIdx.y;
  A += batch * aBatch; W += batch * wBatch;
  bias += batch * biasBatch; out += batch * outBatch;
  const int mTile = blockIdx.x * 128;

  v8f acc[16];
#pragma unroll
  for (int nt = 0; nt < 16; ++nt) {
    float bv = bias[nt * 16 + lanelo];
#pragma unroll
    for (int r = 0; r < 8; ++r) acc[nt][r] = bv;
  }

  for (int kc = 0; kc < K; kc += 64) {
#ifdef HAVE_TDM
    if (wv == 0) {
      // W chunk 256x64: pad 16B after each 128B row (codes 4,3)
      tdm_load_2d((unsigned)(size_t)ldsW, W + kc, 256, 64, (unsigned long long)K, 4, 3);
      // A tiles 128x32: pad 8B after each 64B row (codes 3,1)
      tdm_load_2d((unsigned)(size_t)ldsA0, A + (long long)mTile * K + kc,
                  128, 32, (unsigned long long)K, 3, 1);
      tdm_load_2d((unsigned)(size_t)ldsA1, A + (long long)mTile * K + kc + 32,
                  128, 32, (unsigned long long)K, 3, 1);
      wait_tensorcnt1();  // W + A0 resident; A1 still in flight
    }
#else
    {
      for (int i = tid; i < 256 * 32; i += 256) {
        int n = i >> 5, cu = i & 31;
        *((unsigned*)(ldsW + n * WPAD) + cu) =
            *(const unsigned*)(W + (long long)n * K + kc + cu * 2);
        if (kc + 64 < K)
          __builtin_prefetch(W + (long long)n * K + kc + 64 + cu * 2, 0, 0);
      }
      for (int i = tid; i < 128 * 16; i += 256) {
        int r = i >> 4, cu = i & 15;
        *((unsigned*)(ldsA0 + r * APAD) + cu) =
            *(const unsigned*)(A + (long long)(mTile + r) * K + kc + cu * 2);
        *((unsigned*)(ldsA1 + r * APAD) + cu) =
            *(const unsigned*)(A + (long long)(mTile + r) * K + kc + 32 + cu * 2);
      }
    }
#endif
    __syncthreads();
    gemm_kstep(ldsA0, ldsW, 0, wv, lanelo, laneHi, acc);
#ifdef HAVE_TDM
    if (wv == 0) wait_tensorcnt0();  // A1 resident before barrier releases
#endif
    __syncthreads();
    gemm_kstep(ldsA1, ldsW, 32, wv, lanelo, laneHi, acc);
    __syncthreads();
  }
#pragma unroll
  for (int nt = 0; nt < 16; ++nt) {
    int n = nt * 16 + lanelo;
#pragma unroll
    for (int r = 0; r < 8; ++r) {
      long long m = mTile + wv * 16 + r + laneHi * 8;
      out[m * 256 + n] = acc[nt][r];
    }
  }
}

// ---------------------------------------------------------------------------
// Shared-weight LSTM scan, one 8-wave workgroup. Per step:
// g = xp[:,t,:] + h(32x64)@w_hh^T(64x256) + b_hh via 8 WMMA/wave with
// register-resident B-fragments; xp for step t+1 prefetched into registers
// during the WMMA/gate phases. LDS region0 stages w_hh, then becomes ldsG.
// ---------------------------------------------------------------------------
__global__ __launch_bounds__(256)
void k_scan_shared(const float* __restrict__ xp, const __bf16* __restrict__ whh,
                   const float* __restrict__ bhh,
                   const float* __restrict__ h0, int h0s,
                   const float* __restrict__ c0, int c0s,
                   float* __restrict__ hout, __bf16* __restrict__ houtb) {
  extern __shared__ char smem[];
  __bf16* ldsW = (__bf16*)smem;                  // 256*64 bf16 (staging only)
  float*  ldsG = (float*)smem;                   // 32*256 f32 (overlay)
  __bf16* ldsH = (__bf16*)(smem + 32 * 256 * 4); // 32 x WPAD bf16

  const int tid = threadIdx.x, lane = tid & 31, wv = tid >> 5;
  const int laneHi = lane >> 4, lanelo = lane & 15;
  const int gb = tid >> 3, gj = (tid & 7) * 8;

  for (int i = tid; i < 256 * 32; i += 256)
    ((unsigned*)ldsW)[i] = ((const unsigned*)whh)[i];
  float c[8];
#pragma unroll
  for (int u = 0; u < 8; ++u) {
    c[u] = c0[gb * c0s + gj + u];
    ldsH[gb * WPAD + gj + u] = (__bf16)h0[gb * h0s + gj + u];
  }
  __syncthreads();

  union BV { v16bf v; u32x4 q[2]; };
  BV bF[2][2];
#pragma unroll
  for (int ks = 0; ks < 2; ++ks)
#pragma unroll
    for (int nt = 0; nt < 2; ++nt) {
      int n = (wv * 2 + nt) * 16 + lanelo;
      const __bf16* wr = ldsW + n * 64 + ks * 32 + laneHi * 16;
      bF[ks][nt].q[0] = *(const u32x4*)wr;
      bF[ks][nt].q[1] = *(const u32x4*)(wr + 8);
    }
  float bias[2] = { bhh[(wv * 2 + 0) * 16 + lanelo], bhh[(wv * 2 + 1) * 16 + lanelo] };
  __syncthreads();  // fragments built before ldsG overlays ldsW

  // software-pipelined xp: registers hold step t's tile
  float xpr[2][2][8];
#pragma unroll
  for (int mt = 0; mt < 2; ++mt)
#pragma unroll
    for (int nt = 0; nt < 2; ++nt)
#pragma unroll
      for (int r = 0; r < 8; ++r) {
        int b = mt * 16 + r + laneHi * 8;
        int n = (wv * 2 + nt) * 16 + lanelo;
        xpr[mt][nt][r] = xp[(long long)b * Tt * 256 + n];
      }

#pragma unroll 1
  for (int t = 0; t < Tt; ++t) {
    v8f acc_[2][2];
#pragma unroll
    for (int mt = 0; mt < 2; ++mt)
#pragma unroll
      for (int nt = 0; nt < 2; ++nt)
#pragma unroll
        for (int r = 0; r < 8; ++r) acc_[mt][nt][r] = xpr[mt][nt][r] + bias[nt];
    if (t + 1 < Tt) {  // prefetch next step's xp (hidden behind wmma+gates)
#pragma unroll
      for (int mt = 0; mt < 2; ++mt)
#pragma unroll
        for (int nt = 0; nt < 2; ++nt)
#pragma unroll
          for (int r = 0; r < 8; ++r) {
            int b = mt * 16 + r + laneHi * 8;
            int n = (wv * 2 + nt) * 16 + lanelo;
            xpr[mt][nt][r] = xp[((long long)b * Tt + t + 1) * 256 + n];
          }
    }
#pragma unroll
    for (int mt = 0; mt < 2; ++mt) {
      union { v16bf v; unsigned u[8]; } aF0, aF1;
      const __bf16* hr = ldsH + (mt * 16 + lanelo) * WPAD;
#pragma unroll
      for (int j = 0; j < 8; ++j) {
        int kk = a_frag_k(j, laneHi);
        aF0.u[j] = *(const unsigned*)(hr + kk);
        aF1.u[j] = *(const unsigned*)(hr + 32 + kk);
      }
#pragma unroll
      for (int nt = 0; nt < 2; ++nt) {
        int n = (wv * 2 + nt) * 16 + lanelo;
        v8f a = acc_[mt][nt];
        a = __builtin_amdgcn_wmma_f32_16x16x32_bf16(false, aF0.v, false, bF[0][nt].v,
                                                    (short)0, a, false, false);
        a = __builtin_amdgcn_wmma_f32_16x16x32_bf16(false, aF1.v, false, bF[1][nt].v,
                                                    (short)0, a, false, false);
#pragma unroll
        for (int r = 0; r < 8; ++r) {
          int b = mt * 16 + r + laneHi * 8;
          ldsG[b * 256 + n] = a[r];
        }
      }
    }
    __syncthreads();
    {
      const float* gbase = ldsG + gb * 256;
      f32x4 I0 = *(const f32x4*)(gbase + gj),       I1 = *(const f32x4*)(gbase + gj + 4);
      f32x4 F0 = *(const f32x4*)(gbase + 64 + gj),  F1 = *(const f32x4*)(gbase + 64 + gj + 4);
      f32x4 G0 = *(const f32x4*)(gbase + 128 + gj), G1 = *(const f32x4*)(gbase + 128 + gj + 4);
      f32x4 O0 = *(const f32x4*)(gbase + 192 + gj), O1 = *(const f32x4*)(gbase + 192 + gj + 4);
      f32x4 hv0, hv1;
      bf16x8 hb;
#pragma unroll
      for (int u = 0; u < 8; ++u) {
        float gi = (u < 4) ? I0[u & 3] : I1[u & 3];
        float gf = (u < 4) ? F0[u & 3] : F1[u & 3];
        float gg = (u < 4) ? G0[u & 3] : G1[u & 3];
        float go = (u < 4) ? O0[u & 3] : O1[u & 3];
        float cc = sigf(gf) * c[u] + sigf(gi) * tanhf(gg);
        c[u] = cc;
        float h = sigf(go) * tanhf(cc);
        if (u < 4) hv0[u & 3] = h; else hv1[u & 3] = h;
        hb[u] = (__bf16)h;
      }
      long long o = ((long long)gb * Tt + t) * 64 + gj;
      *(f32x4*)(hout + o) = hv0;
      *(f32x4*)(hout + o + 4) = hv1;
      *(bf16x8*)(houtb + o) = hb;
      *(bf16x8*)(ldsH + gb * WPAD + gj) = hb;
    }
    __syncthreads();
  }
}

// ---------------------------------------------------------------------------
// Per-sample LSTM scan: one workgroup per sample, modulated w_hh_eff[b]
// LDS-resident (transposed bf16, conflict-free), xp prefetched one step ahead.
// ---------------------------------------------------------------------------
__global__ __launch_bounds__(256)
void k_scan_persample(const float* __restrict__ xp, const float* __restrict__ whh,
                      const float* __restrict__ bhh,
                      const float* __restrict__ h0, int h0s,
                      const float* __restrict__ c0, int c0s,
                      float* __restrict__ hout, __bf16* __restrict__ houtb) {
  extern __shared__ char smem[];
  __bf16* ldsWt = (__bf16*)smem;                    // [64][256] transposed
  float*  ldsH  = (float*)(smem + 64 * 256 * 2);    // [64]
  float*  ldsG  = ldsH + 64;                        // [256]
  const int b = blockIdx.x, tid = threadIdx.x;

  for (int i = tid; i < 256 * 64; i += 256) {
    int n = i >> 6, ii = i & 63;
    ldsWt[ii * 256 + n] = (__bf16)whh[((long long)b * 256 + n) * 64 + ii];
  }
  if (tid < 64) ldsH[tid] = h0[b * h0s + tid];
  float c_ = (tid < 64) ? c0[b * c0s + tid] : 0.0f;
  const float bb = bhh[b * 256 + tid];
  __syncthreads();

  const float* xpb = xp + (long long)b * Tt * 256;
  float xv = xpb[tid];
#pragma unroll 1
  for (int t = 0; t < Tt; ++t) {
    float g = xv + bb;
    if (t + 1 < Tt) xv = xpb[(long long)(t + 1) * 256 + tid];
#pragma unroll 8
    for (int i = 0; i < 64; ++i) g += ldsH[i] * (float)ldsWt[i * 256 + tid];
    ldsG[tid] = g;
    __syncthreads();
    if (tid < 64) {
      float gi = ldsG[tid], gf = ldsG[64 + tid], gg = ldsG[128 + tid], go = ldsG[192 + tid];
      c_ = sigf(gf) * c_ + sigf(gi) * tanhf(gg);
      float h = sigf(go) * tanhf(c_);
      long long o = ((long long)b * Tt + t) * 64 + tid;
      hout[o] = h;
      houtb[o] = (__bf16)h;
      ldsH[tid] = h;
    }
    __syncthreads();
  }
}

// ---------------------------------------------------------------------------
// Hypernet conv1d(k=3, SAME) + BN + ReLU, direct f32.
// ---------------------------------------------------------------------------
__global__ void k_conv_bn_relu(const float* __restrict__ x, const float* __restrict__ style,
                               int Cin, int L,
                               const float* __restrict__ w, const float* __restrict__ gamma,
                               const float* __restrict__ beta, float* __restrict__ y, int Cout) {
  int kb = blockIdx.x;
  int k = kb >> 5, bsm = kb & 31;
  int idx = blockIdx.y * 256 + threadIdx.x;
  if (idx >= Cout * L) return;
  int co = idx / L, pos = idx % L;
  const float* wp = w + ((long long)(k * Cout + co)) * Cin * 3;
  float acc = 0.0f;
  if (style) {
    const float* xr = style + bsm * 256;
    float xm = (pos > 0) ? xr[pos - 1] : 0.0f;
    float x0 = xr[pos];
    float xq = (pos < L - 1) ? xr[pos + 1] : 0.0f;
    acc = xm * wp[0] + x0 * wp[1] + xq * wp[2];
  } else {
    const float* xb = x + (long long)kb * Cin * L;
    for (int ci = 0; ci < Cin; ++ci) {
      const float* xr = xb + ci * L;
      const float* wc = wp + ci * 3;
      float xm = (pos > 0) ? xr[pos - 1] : 0.0f;
      float x0 = xr[pos];
      float xq = (pos < L - 1) ? xr[pos + 1] : 0.0f;
      acc += xm * wc[0] + x0 * wc[1] + xq * wc[2];
    }
  }
  float sc = gamma[k * Cout + co] * rsqrtf(1.0f + 1e-5f);
  float v = acc * sc + beta[k * Cout + co];
  y[(long long)kb * Cout * L + idx] = v > 0.0f ? v : 0.0f;
}

__global__ void k_maxpool2(const float* __restrict__ x, float* __restrict__ y,
                           int C, int Lout) {
  long long idx = (long long)blockIdx.x * 256 + threadIdx.x;
  long long total = 256LL * C * Lout;
  if (idx >= total) return;
  int kb = (int)(idx / (C * Lout));
  int rem = (int)(idx % (C * Lout));
  int cch = rem / Lout, p = rem % Lout;
  const float* xr = x + ((long long)kb * C + cch) * (2 * Lout) + 2 * p;
  float a = xr[0], bval = xr[1];
  y[idx] = a > bval ? a : bval;
}

__global__ void k_pb(const float* __restrict__ style, const float* __restrict__ Wb,
                     const float* __restrict__ bbv, float* __restrict__ pb) {
  int idx = blockIdx.x * 256 + threadIdx.x;  // 8*32*256
  int k = idx >> 13, b = (idx >> 8) & 31, g = idx & 255;
  const float* s = style + b * 256;
  const float* wr = Wb + ((long long)k * 256 + g) * 256;
  float acc = bbv[k * 256 + g];
  for (int ss = 0; ss < 256; ++ss) acc += s[ss] * wr[ss];
  pb[idx] = acc;
}

__global__ void k_h0c0(const float* __restrict__ style,
                       const float* __restrict__ Wh, const float* __restrict__ bh,
                       const float* __restrict__ Wc, const float* __restrict__ bc,
                       float* __restrict__ h0, float* __restrict__ c0) {
  int idx = blockIdx.x * 256 + threadIdx.x;  // 32*320
  if (idx >= 32 * 320) return;
  int b = idx / 320, j = idx % 320;
  const float* s = style + b * 256;
  const float* whr = Wh + (long long)j * 256;
  const float* wcr = Wc + (long long)j * 256;
  float ah = bh[j], ac = bc[j];
  for (int ss = 0; ss < 256; ++ss) { ah += s[ss] * whr[ss]; ac += s[ss] * wcr[ss]; }
  h0[idx] = ah;
  c0[idx] = ac;
}

__global__ void k_cvt_bf16(const float* __restrict__ x, __bf16* __restrict__ y, long long n) {
  long long idx = (long long)blockIdx.x * 256 + threadIdx.x;
  if (idx < n) y[idx] = (__bf16)x[idx];
}

__global__ void k_build_xin(const float* __restrict__ latent, const float* __restrict__ style,
                            __bf16* __restrict__ xin) {
  long long idx = (long long)blockIdx.x * 256 + threadIdx.x;
  if (idx >= (long long)Bt * Tt * 320) return;
  int col = (int)(idx % 320);
  long long bt = idx / 320;
  int b = (int)(bt / Tt);
  float v = (col < 64) ? latent[bt * 64 + col] : style[b * 256 + (col - 64)];
  xin[idx] = (__bf16)v;
}

__global__ void k_build_eff(const float* __restrict__ ra_w_ih, const float* __restrict__ ra_w_hh,
                            const float* __restrict__ ra_b_ih, const float* __restrict__ ra_b_hh,
                            const float* __restrict__ hyp, const float* __restrict__ pb, int k,
                            __bf16* __restrict__ wih, float* __restrict__ whh,
                            float* __restrict__ bih, float* __restrict__ bhh) {
  int idx = blockIdx.x * 256 + threadIdx.x;  // B*G*H
  if (idx < 32 * 256 * 64) {
    int i = idx & 63, g = (idx >> 6) & 255, b = idx >> 14;
    float wi = ra_w_ih[(k * 256 + g) * 64 + i];
    float wh = ra_w_hh[(k * 256 + g) * 64 + i];
    long long hI = (((long long)(k * 32 + b)) * 256 + g) * 64 + i;
    long long hI2 = (((long long)((4 + k) * 32 + b)) * 256 + g) * 64 + i;
    wih[idx] = (__bf16)(wi * (1.0f + hyp[hI]));
    whh[idx] = wh * (1.0f + hyp[hI2]);
  }
  if (idx < 32 * 256) {
    int g = idx & 255, b = idx >> 8;
    bih[idx] = ra_b_ih[k * 256 + g] * (1.0f + pb[(k * 32 + b) * 256 + g]);
    bhh[idx] = ra_b_hh[k * 256 + g] * (1.0f + pb[((4 + k) * 32 + b) * 256 + g]);
  }
}

__global__ void k_add(const float* __restrict__ a, const float* __restrict__ b,
                      float* __restrict__ o, long long n) {
  long long idx = (long long)blockIdx.x * 256 + threadIdx.x;
  if (idx < n) o[idx] = a[idx] + b[idx];
}

// ---------------------------------------------------------------------------
static inline long long ceilDiv(long long a, long long b) { return (a + b - 1) / b; }

extern "C" void kernel_launch(void* const* d_in, const int* in_sizes, int n_in,
                              void* d_out, int out_size, void* d_ws, size_t ws_size,
                              hipStream_t stream) {
  const float* latent   = (const float*)d_in[0];
  const float* style    = (const float*)d_in[1];
  const float* n_w_ih   = (const float*)d_in[2];
  const float* n_w_hh   = (const float*)d_in[3];
  const float* n_b_ih   = (const float*)d_in[4];
  const float* n_b_hh   = (const float*)d_in[5];
  const float* n_h0     = (const float*)d_in[6];
  const float* n_c0     = (const float*)d_in[7];
  const float* ra_w_ih0 = (const float*)d_in[8];
  const float* ra_w_hh0 = (const float*)d_in[9];
  const float* ra_b_ih0 = (const float*)d_in[10];
  const float* ra_b_hh0 = (const float*)d_in[11];
  const float* ra_w_ih  = (const float*)d_in[12];
  const float* ra_w_hh  = (const float*)d_in[13];
  const float* ra_b_ih  = (const float*)d_in[14];
  const float* ra_b_hh  = (const float*)d_in[15];
  const float* W_h0 = (const float*)d_in[16];
  const float* b_h0 = (const float*)d_in[17];
  const float* W_c0 = (const float*)d_in[18];
  const float* b_c0 = (const float*)d_in[19];
  const float* hcv[6] = {(const float*)d_in[20], (const float*)d_in[23], (const float*)d_in[26],
                         (const float*)d_in[29], (const float*)d_in[32], (const float*)d_in[35]};
  const float* hgv[6] = {(const float*)d_in[21], (const float*)d_in[24], (const float*)d_in[27],
                         (const float*)d_in[30], (const float*)d_in[33], (const float*)d_in[36]};
  const float* hbv[6] = {(const float*)d_in[22], (const float*)d_in[25], (const float*)d_in[28],
                         (const float*)d_in[31], (const float*)d_in[34], (const float*)d_in[37]};
  const float* Wb  = (const float*)d_in[38];
  const float* bbv = (const float*)d_in[39];

  const long long Mall = (long long)Bt * Tt;  // 65536
  char* ws = (char*)d_ws;
  size_t off = 0;
  auto alloc = [&](size_t bytes) { char* p = ws + off; off = (off + bytes + 255) & ~(size_t)255; return p; };

  float*  ws_xp       = (float*)alloc(Mall * Gt * 4);     // 64 MB (also conv ping-pong)
  float*  ws_hA       = (float*)alloc(Mall * Ht * 4);
  __bf16* ws_hA_bf    = (__bf16*)alloc(Mall * Ht * 2);
  float*  ws_hB       = (float*)alloc(Mall * Ht * 4);
  __bf16* ws_hB_bf    = (__bf16*)alloc(Mall * Ht * 2);
  float*  ws_neutral  = (float*)alloc(Mall * Ht * 4);
  __bf16* ws_lat_bf   = (__bf16*)alloc(Mall * Ht * 2);
  __bf16* ws_xin      = (__bf16*)alloc(Mall * 320 * 2);
  float*  ws_hyp      = (float*)alloc(256LL * 256 * 64 * 4);
  float*  ws_pb       = (float*)alloc(8LL * 32 * 256 * 4);
  float*  ws_h0       = (float*)alloc(32LL * 320 * 4);
  float*  ws_c0       = (float*)alloc(32LL * 320 * 4);
  __bf16* ws_wih_eff  = (__bf16*)alloc(32LL * 256 * 64 * 2);
  float*  ws_whh_eff  = (float*)alloc(32LL * 256 * 64 * 4);
  float*  ws_bih_eff  = (float*)alloc(32LL * 256 * 4);
  float*  ws_bhh_eff  = (float*)alloc(32LL * 256 * 4);
  __bf16* ws_nwih_bf  = (__bf16*)alloc(2LL * 256 * 64 * 2);
  __bf16* ws_nwhh_bf  = (__bf16*)alloc(2LL * 256 * 64 * 2);
  __bf16* ws_rwih0_bf = (__bf16*)alloc(256LL * 320 * 2);
  __bf16* ws_rwhh0_bf = (__bf16*)alloc(256LL * 64 * 2);
  float* bufA = ws_xp;
  float* bufB = ws_xp + (16LL << 20) / 4;
  (void)in_sizes; (void)n_in; (void)out_size; (void)ws_size;

  const size_t shGemm  = 256 * WPAD * 2 + 2 * 128 * APAD * 2;   // 54 KB
  const size_t shScanS = 32 * 256 * 4 + 32 * WPAD * 2;          // ~37 KB
  const size_t shScanP = 64 * 256 * 2 + 64 * 4 + 256 * 4;       // ~34 KB

  k_cvt_bf16<<<(unsigned)ceilDiv(Mall * Ht, 256), 256, 0, stream>>>(latent, ws_lat_bf, Mall * Ht);
  k_cvt_bf16<<<(unsigned)ceilDiv(2 * 256 * 64, 256), 256, 0, stream>>>(n_w_ih, ws_nwih_bf, 2 * 256 * 64);
  k_cvt_bf16<<<(unsigned)ceilDiv(2 * 256 * 64, 256), 256, 0, stream>>>(n_w_hh, ws_nwhh_bf, 2 * 256 * 64);
  k_cvt_bf16<<<(unsigned)ceilDiv(256 * 320, 256), 256, 0, stream>>>(ra_w_ih0, ws_rwih0_bf, 256 * 320);
  k_cvt_bf16<<<(unsigned)ceilDiv(256 * 64, 256), 256, 0, stream>>>(ra_w_hh0, ws_rwhh0_bf, 256 * 64);

  k_conv_bn_relu<<<dim3(256, 64), 256, 0, stream>>>(nullptr, style, 1, 256, hcv[0], hgv[0], hbv[0], bufA, 64);
  k_conv_bn_relu<<<dim3(256, 64), 256, 0, stream>>>(bufA, nullptr, 64, 256, hcv[1], hgv[1], hbv[1], bufB, 64);
  k_maxpool2<<<(unsigned)ceilDiv(256LL * 64 * 128, 256), 256, 0, stream>>>(bufB, bufA, 64, 128);
  k_conv_bn_relu<<<dim3(256, 64), 256, 0, stream>>>(bufA, nullptr, 64, 128, hcv[2], hgv[2], hbv[2], bufB, 128);
  k_conv_bn_relu<<<dim3(256, 64), 256, 0, stream>>>(bufB, nullptr, 128, 128, hcv[3], hgv[3], hbv[3], bufA, 128);
  k_maxpool2<<<(unsigned)ceilDiv(256LL * 128 * 64, 256), 256, 0, stream>>>(bufA, bufB, 128, 64);
  k_conv_bn_relu<<<dim3(256, 64), 256, 0, stream>>>(bufB, nullptr, 128, 64, hcv[4], hgv[4], hbv[4], bufA, 256);
  k_conv_bn_relu<<<dim3(256, 64), 256, 0, stream>>>(bufA, nullptr, 256, 64, hcv[5], hgv[5], hbv[5], ws_hyp, 256);

  k_pb<<<256, 256, 0, stream>>>(style, Wb, bbv, ws_pb);
  k_h0c0<<<40, 256, 0, stream>>>(style, W_h0, b_h0, W_c0, b_c0, ws_h0, ws_c0);

  // --- neutral path ---
  k_gemm_bf16<<<dim3(512, 1), 256, shGemm, stream>>>(ws_lat_bf, 0, ws_nwih_bf, 0,
                                                     n_b_ih, 0, ws_xp, 0, (int)Mall, 64);
  k_scan_shared<<<1, 256, shScanS, stream>>>(ws_xp, ws_nwhh_bf, n_b_hh,
                                             n_h0, 0, n_c0, 0, ws_hA, ws_hA_bf);
  k_gemm_bf16<<<dim3(512, 1), 256, shGemm, stream>>>(ws_hA_bf, 0, ws_nwih_bf + 256 * 64, 0,
                                                     n_b_ih + 256, 0, ws_xp, 0, (int)Mall, 64);
  k_scan_shared<<<1, 256, shScanS, stream>>>(ws_xp, ws_nwhh_bf + 256 * 64, n_b_hh + 256,
                                             n_h0 + 64, 0, n_c0 + 64, 0, ws_neutral, ws_hB_bf);

  // --- residual adapter shared layer 0 (K = 320) ---
  k_build_xin<<<(unsigned)ceilDiv(Mall * 320, 256), 256, 0, stream>>>(latent, style, ws_xin);
  k_gemm_bf16<<<dim3(512, 1), 256, shGemm, stream>>>(ws_xin, 0, ws_rwih0_bf, 0,
                                                     ra_b_ih0, 0, ws_xp, 0, (int)Mall, 320);
  k_scan_shared<<<1, 256, shScanS, stream>>>(ws_xp, ws_rwhh0_bf, ra_b_hh0,
                                             ws_h0, 320, ws_c0, 320, ws_hA, ws_hA_bf);

  // --- 4 per-sample hyper-modulated layers ---
  float* curF = ws_hA; __bf16* curB = ws_hA_bf;
  float* nxtF = ws_hB; __bf16* nxtB = ws_hB_bf;
  for (int k = 0; k < 4; ++k) {
    k_build_eff<<<2048, 256, 0, stream>>>(ra_w_ih, ra_w_hh, ra_b_ih, ra_b_hh,
                                          ws_hyp, ws_pb, k,
                                          ws_wih_eff, ws_whh_eff, ws_bih_eff, ws_bhh_eff);
    k_gemm_bf16<<<dim3(16, 32), 256, shGemm, stream>>>(curB, (long long)Tt * Ht,
                                                       ws_wih_eff, (long long)Gt * Ht,
                                                       ws_bih_eff, (long long)Gt,
                                                       ws_xp, (long long)Tt * Gt, Tt, 64);
    k_scan_persample<<<32, 256, shScanP, stream>>>(ws_xp, ws_whh_eff, ws_bhh_eff,
                                                   ws_h0 + (k + 1) * 64, 320,
                                                   ws_c0 + (k + 1) * 64, 320,
                                                   nxtF, nxtB);
    float* tf = curF; curF = nxtF; nxtF = tf;
    __bf16* tb = curB; curB = nxtB; nxtB = tb;
  }

  k_add<<<(unsigned)ceilDiv(Mall * Ht, 256), 256, 0, stream>>>(ws_neutral, curF, (float*)d_out, Mall * Ht);
}
